// Attention_25331717111812
// MI455X (gfx1250) — matrix operand order repaired and checked by reading, not compiled, not run on hardware
//
#include <hip/hip_runtime.h>
#include <hip/hip_bf16.h>

typedef __bf16 bf16_t;
typedef __attribute__((ext_vector_type(16))) __bf16 v16bf;
typedef __attribute__((ext_vector_type(8)))  float  v8f;

#define BATCH 8
#define SEQ   2048
#define HD    1024
#define BS    (BATCH * SEQ)

// ---------- WMMA fragment helpers (CDNA5 wave32 16x16x32 bf16 layouts) ----------
// A-matrix 16x32: lane L holds row M=L%16; half h=L/16:
//   elems 0..7  -> K = h*8 + e        (contiguous 8)
//   elems 8..15 -> K = 16 + h*8 + e-8 (contiguous 8)
__device__ __forceinline__ v16bf load_a16(const bf16_t* __restrict__ rowbase, int h) {
    v16bf a;
    const bf16_t* p0 = rowbase + h * 8;
    const bf16_t* p1 = rowbase + 16 + h * 8;
#pragma unroll
    for (int i = 0; i < 8; ++i) { a[i] = p0[i]; a[8 + i] = p1[i]; }
    return a;
}
// B-matrix 32x16: lane L holds col N=L%16; half h: elems 0..7 -> K = h*8 + e, elems 8..15 -> K = 16 + h*8 + e-8
__device__ __forceinline__ v16bf load_b16(const bf16_t* __restrict__ colbase, int h) {
    v16bf b;
    const bf16_t* p0 = colbase + h * 8;
    const bf16_t* p1 = colbase + 16 + h * 8;
#pragma unroll
    for (int i = 0; i < 8; ++i) { b[i] = p0[i]; b[8 + i] = p1[i]; }
    return b;
}
__device__ __forceinline__ v8f wmma_bf16(v16bf a, v16bf b, v8f c) {
    return __builtin_amdgcn_wmma_f32_16x16x32_bf16(false, a, false, b, (short)0, c, false, false);
}

// ------------------------------------------------------------------
// Kernel 1: fused QKV projection.  out[m,d] = relu(X[m,:]·W[d,:] + b[d])
// grid = (HD/128, BS/128, 3); z selects Q/K/V.  V stored transposed: Vt[b][d][s].
// Staged K-depth 64 -> 16 WMMAs per barrier pair.
// ------------------------------------------------------------------
__global__ __launch_bounds__(256) void qkv_gemm(
    const float* __restrict__ X,
    const float* __restrict__ Wq, const float* __restrict__ bq,
    const float* __restrict__ Wk, const float* __restrict__ bk,
    const float* __restrict__ Wv, const float* __restrict__ bv,
    bf16_t* __restrict__ Qb, bf16_t* __restrict__ Kb, bf16_t* __restrict__ Vt)
{
    const int z = blockIdx.z;
    const float* W    = (z == 0) ? Wq : (z == 1) ? Wk : Wv;
    const float* bias = (z == 0) ? bq : (z == 1) ? bk : bv;

    const int m0 = blockIdx.y * 128;
    const int n0 = blockIdx.x * 128;
    const int tid = threadIdx.x;
    const int w = tid >> 5, lane = tid & 31, h = lane >> 4, ln = lane & 15;
    const int wm = w >> 1;          // 0..3: 32-row strip
    const int wn = w & 1;           // 0..1: 64-col strip

    __shared__ bf16_t Asm[128][72];  // 128 x 64 (+8 pad), X tile in bf16
    __shared__ bf16_t Bsm[128][72];  // 128 x 64 (+8 pad), W tile (row d, col h)

    v8f acc[2][4];
#pragma unroll
    for (int i = 0; i < 2; ++i)
#pragma unroll
        for (int j = 0; j < 4; ++j)
#pragma unroll
            for (int r = 0; r < 8; ++r) acc[i][j][r] = 0.0f;

    const int lrow = tid >> 1;            // 0..127
    const int lcb  = (tid & 1) * 32;      // 0 or 32

    for (int kb = 0; kb < HD; kb += 64) {
        const float* sa = X + (size_t)(m0 + lrow) * HD + kb + lcb;
        const float* sb = W + (size_t)(n0 + lrow) * HD + kb + lcb;
        if (kb + 64 < HD) {
            __builtin_prefetch(sa + 64, 0, 1);
            __builtin_prefetch(sb + 64, 0, 1);
        }
#pragma unroll
        for (int i = 0; i < 32; ++i) Asm[lrow][lcb + i] = (bf16_t)sa[i];
#pragma unroll
        for (int i = 0; i < 32; ++i) Bsm[lrow][lcb + i] = (bf16_t)sb[i];
        __syncthreads();

#pragma unroll
        for (int kk = 0; kk < 64; kk += 32) {
            v16bf a0 = load_a16(&Asm[wm * 32 +      ln][kk], h);
            v16bf a1 = load_a16(&Asm[wm * 32 + 16 + ln][kk], h);
#pragma unroll
            for (int nt = 0; nt < 4; ++nt) {
                v16bf bfr = load_b16(&Bsm[wn * 64 + nt * 16 + ln][kk], h);
                acc[0][nt] = wmma_bf16(a0, bfr, acc[0][nt]);
                acc[1][nt] = wmma_bf16(a1, bfr, acc[1][nt]);
            }
        }
        __syncthreads();
    }

#pragma unroll
    for (int mt = 0; mt < 2; ++mt)
#pragma unroll
        for (int nt = 0; nt < 4; ++nt) {
            const int d = n0 + wn * 64 + nt * 16 + ln;
            const float bv_ = bias[d];
#pragma unroll
            for (int r = 0; r < 8; ++r) {
                const int m = m0 + wm * 32 + mt * 16 + r + 8 * h;
                float v = acc[mt][nt][r] + bv_;
                v = v > 0.0f ? v : 0.0f;
                const bf16_t o = (bf16_t)v;
                if (z == 0)      Qb[(size_t)m * HD + d] = o;
                else if (z == 1) Kb[(size_t)m * HD + d] = o;
                else {
                    const int b = m >> 11, s = m & 2047;
                    Vt[((size_t)b * HD + d) * SEQ + s] = o;
                }
            }
        }
}

// ------------------------------------------------------------------
// Kernel 2: fused flash attention.
// block = 32 q-rows; 8 waves = 2 row-groups(16) x 4 d-slices(256).
// Online softmax over k-chunks of 64; out = (P·V)/l + X residual.
// Per chunk each wave: 32 score WMMAs (2 k-tiles over its 512-d half)
// + 32 O WMMAs; 2 block barriers.
// ------------------------------------------------------------------
__global__ __launch_bounds__(256) void attn_flash(
    const bf16_t* __restrict__ Qb, const bf16_t* __restrict__ Kb,
    const bf16_t* __restrict__ Vt,
    const float* __restrict__ X, const float* __restrict__ mask,
    float* __restrict__ out)
{
    const int blk = blockIdx.x;
    const int b   = blk >> 6;          // SEQ/32 = 64 q-tiles per batch
    const int q0  = (blk & 63) * 32;   // q-row base within batch
    const int tid = threadIdx.x, w = tid >> 5, lane = tid & 31;
    const int h = lane >> 4, ln = lane & 15;
    const int rg    = w >> 2;          // row-group 0/1 (16 rows each)
    const int ds    = w & 3;           // d-slice 0..3 (256 cols each)
    const int dh    = ds & 1;          // d-half (512) for score partials
    const int tbase = ds >> 1;         // this wave covers k-tiles {tbase, tbase+2}

    const bf16_t* Qbase = Qb + ((size_t)b * SEQ + q0 + rg * 16) * HD;
    const bf16_t* Kbase = Kb + (size_t)b * SEQ * HD;
    const bf16_t* Vbase = Vt + (size_t)b * HD * SEQ;

    __shared__ float  Sp[2][8][16][16];   // score partials: [rg][tile*2+half]
    __shared__ bf16_t Pb[2][16][68];      // probabilities (row-major 16x64, padded)
    __shared__ float  Scl[2][16];         // per-row O rescale factor for this chunk
    __shared__ float  Inv[2][16];         // per-row 1/l at the end

    // Q A-fragments: 16 rows x 512 d (this wave's half), resident in VGPRs.
    v16bf qa[16];
#pragma unroll
    for (int i = 0; i < 16; ++i)
        qa[i] = load_a16(Qbase + (size_t)ln * HD + dh * 512 + i * 32, h);

    // O accumulators: 16 rows x 256 d-slice = 16 C-fragments.
    v8f o[16];
#pragma unroll
    for (int nt = 0; nt < 16; ++nt)
#pragma unroll
        for (int r = 0; r < 8; ++r) o[nt][r] = 0.0f;

    // Online softmax state (valid in ds==0 waves): lane row = ln, col-half = h.
    float mrow = -3.0e38f, lrow = 0.0f;

    for (int kc = 0; kc < SEQ; kc += 64) {
        // ---- partial scores: two 16x16 tiles over this wave's d-half ----
        v8f sp0, sp1;
#pragma unroll
        for (int r = 0; r < 8; ++r) { sp0[r] = 0.0f; sp1[r] = 0.0f; }
        const bf16_t* k0 = Kbase + (size_t)(kc + tbase * 16      + ln) * HD + dh * 512;
        const bf16_t* k1 = Kbase + (size_t)(kc + tbase * 16 + 32 + ln) * HD + dh * 512;
#pragma unroll
        for (int i = 0; i < 16; ++i) sp0 = wmma_bf16(qa[i], load_b16(k0 + i * 32, h), sp0);
#pragma unroll
        for (int i = 0; i < 16; ++i) sp1 = wmma_bf16(qa[i], load_b16(k1 + i * 32, h), sp1);
        if (kc + 64 < SEQ) {  // prefetch next K chunk (global_prefetch_b8)
            __builtin_prefetch(k0 + 64 * HD, 0, 1);
            __builtin_prefetch(k1 + 64 * HD, 0, 1);
        }
#pragma unroll
        for (int r = 0; r < 8; ++r) {
            Sp[rg][ tbase      * 2 + dh][r + 8 * h][ln] = sp0[r];
            Sp[rg][(tbase + 2) * 2 + dh][r + 8 * h][ln] = sp1[r];
        }
        __syncthreads();

        // ---- online softmax: one wave per row-group; lane: row=ln, cols h*32.. ----
        if (ds == 0) {
            float s[32];
#pragma unroll
            for (int c = 0; c < 32; ++c) {
                const int col = h * 32 + c;
                const int t = col >> 4, cc = col & 15;
                float v = Sp[rg][t * 2][ln][cc] + Sp[rg][t * 2 + 1][ln][cc];
                float mk = mask[(size_t)b * SEQ + kc + col];
                s[c] = v - 1.0e10f * (1.0f - mk);
            }
            float cmax = s[0];
#pragma unroll
            for (int c = 1; c < 32; ++c) cmax = fmaxf(cmax, s[c]);
            cmax = fmaxf(cmax, __shfl_xor(cmax, 16, 32));    // join col halves
            const float mnew  = fmaxf(mrow, cmax);
            const float scale = __expf(mrow - mnew);
            float ls = 0.0f;
#pragma unroll
            for (int c = 0; c < 32; ++c) {
                float e = __expf(s[c] - mnew);
                ls += e;
                Pb[rg][ln][h * 32 + c] = (bf16_t)e;
            }
            ls += __shfl_xor(ls, 16, 32);
            lrow = lrow * scale + ls;
            mrow = mnew;
            if (h == 0) Scl[rg][ln] = scale;
        }
        __syncthreads();

        // ---- rescale O, accumulate P(16x64) · V(64x256-slice) ----
        float sc[8];
#pragma unroll
        for (int r = 0; r < 8; ++r) sc[r] = Scl[rg][r + 8 * h];
#pragma unroll
        for (int nt = 0; nt < 16; ++nt)
#pragma unroll
            for (int r = 0; r < 8; ++r) o[nt][r] *= sc[r];

        v16bf pa0 = load_a16(&Pb[rg][ln][0],  h);
        v16bf pa1 = load_a16(&Pb[rg][ln][32], h);
#pragma unroll
        for (int nt = 0; nt < 16; ++nt) {
            const bf16_t* vp = Vbase + (size_t)(ds * 256 + nt * 16 + ln) * SEQ + kc;
            o[nt] = wmma_bf16(pa0, load_b16(vp,      h), o[nt]);
            o[nt] = wmma_bf16(pa1, load_b16(vp + 32, h), o[nt]);
        }
        if (kc + 64 < SEQ)   // prefetch next V chunk
            __builtin_prefetch(Vbase + (size_t)(ds * 256 + ln) * SEQ + kc + 64, 0, 1);
    }

    // ---- epilogue: O/l + residual ----
    if (ds == 0 && h == 0) Inv[rg][ln] = 1.0f / lrow;
    __syncthreads();
    float il[8];
#pragma unroll
    for (int r = 0; r < 8; ++r) il[r] = Inv[rg][r + 8 * h];

    const float* Xb = X   + ((size_t)b * SEQ + q0 + rg * 16) * HD;
    float*       Ob = out + ((size_t)b * SEQ + q0 + rg * 16) * HD;
#pragma unroll
    for (int nt = 0; nt < 16; ++nt)
#pragma unroll
        for (int r = 0; r < 8; ++r) {
            const int row = r + 8 * h;
            const int d   = ds * 256 + nt * 16 + ln;
            const size_t idx = (size_t)row * HD + d;
            Ob[idx] = o[nt][r] * il[r] + Xb[idx];
        }
}

// ------------------------------------------------------------------
extern "C" void kernel_launch(void* const* d_in, const int* in_sizes, int n_in,
                              void* d_out, int out_size, void* d_ws, size_t ws_size,
                              hipStream_t stream) {
    const float* X    = (const float*)d_in[0];
    const float* mask = (const float*)d_in[1];
    const float* Wq   = (const float*)d_in[2];
    const float* bq   = (const float*)d_in[3];
    const float* Wk   = (const float*)d_in[4];
    const float* bk   = (const float*)d_in[5];
    const float* Wv   = (const float*)d_in[6];
    const float* bv   = (const float*)d_in[7];
    float* out = (float*)d_out;

    const size_t elems = (size_t)BS * HD;
    bf16_t* Qb = (bf16_t*)d_ws;
    bf16_t* Kb = Qb + elems;
    bf16_t* Vt = Kb + elems;   // stored transposed [b][d][s]

    qkv_gemm<<<dim3(HD / 128, BS / 128, 3), 256, 0, stream>>>(
        X, Wq, bq, Wk, bk, Wv, bv, Qb, Kb, Vt);

    attn_flash<<<dim3(BS / 32), 256, 0, stream>>>(Qb, Kb, Vt, X, mask, out);
}
